// HGT_10170482557467
// MI455X (gfx1250) — matrix operand order repaired and checked by reading, not compiled, not run on hardware
//
#include <hip/hip_runtime.h>
#include <stdint.h>

// ---------------- constants ----------------
#define Hh 4
#define Dd 32
#define Cc 128
#define Ll 2
#define INV_SQRT_Df 0.17677669529663687f  // 1/sqrt(32)

#define GM_RELU 0
#define GM_BIAS 1
#define GM_SKIP 2

typedef __attribute__((ext_vector_type(16))) __bf16 v16bf;
typedef __attribute__((ext_vector_type(8)))  float  v8f;

union Frag {
    v16bf v;
    uint4 q[2];
};

// ---------------- device helpers ----------------
__device__ __forceinline__ unsigned short f2bf(float f) {
    unsigned u = __float_as_uint(f);
    unsigned r = u + 0x7FFFu + ((u >> 16) & 1u);   // round-to-nearest-even
    return (unsigned short)(r >> 16);
}
__device__ __forceinline__ unsigned fkey(float f) {   // monotone float->uint key
    unsigned u = __float_as_uint(f);
    return (u & 0x80000000u) ? ~u : (u | 0x80000000u);
}
__device__ __forceinline__ float fdec(unsigned k) {
    return (k & 0x80000000u) ? __uint_as_float(k & 0x7FFFFFFFu) : __uint_as_float(~k);
}

// ---------------- weight prep ----------------
// WT[j*K + k] = bf16(W[k*N + j])   (W is K x N row-major, N columns)
__global__ __launch_bounds__(256)
void hgt_transpose_bf16(const float* __restrict__ W, unsigned short* __restrict__ WT,
                        int K, int N) {
    int i = blockIdx.x * 256 + threadIdx.x;
    if (i >= K * N) return;
    int j = i % N, k = i / N;
    WT[(size_t)j * K + k] = f2bf(W[i]);
}

// Fold per-head relation matrix into projection:
// WT[j=h*32+e][c] = sum_d W[c][h*32+d] * rel[h][d][e] ; bOut[j] = sum_d b[h*32+d]*rel[h][d][e]
__global__ __launch_bounds__(256)
void hgt_combine_rel(const float* __restrict__ W, const float* __restrict__ b,
                     const float* __restrict__ rel,
                     unsigned short* __restrict__ WT, float* __restrict__ bOut) {
    int i = blockIdx.x * 256 + threadIdx.x;
    if (i >= Cc * Cc) return;
    int c = i & 127, j = i >> 7;
    int h = j >> 5, e = j & 31;
    const float* wr = W + (size_t)c * Cc + h * Dd;       // W[c][h*32+d]
    const float* rr = rel + h * Dd * Dd + e;             // rel[h][d][e], stride 32 in d
    float s = 0.0f;
#pragma unroll
    for (int d = 0; d < Dd; ++d) s += wr[d] * rr[d * Dd];
    WT[(size_t)j * Cc + c] = f2bf(s);
    if (c == 0) {
        const float* br = b + h * Dd;
        float sb = 0.0f;
#pragma unroll
        for (int d = 0; d < Dd; ++d) sb += br[d] * rr[d * Dd];
        bOut[j] = sb;
    }
}

__global__ __launch_bounds__(256)
void hgt_cvt_bf16(const float* __restrict__ x, unsigned short* __restrict__ y, int n) {
    int i = blockIdx.x * 256 + threadIdx.x;
    if (i < n) y[i] = f2bf(x[i]);
}

// ---------------- bf16 WMMA GEMM: out[M x 128] = A[M x K] * W[K x 128] ----------------
// A row-major bf16, WT = W transposed to [col][K] bf16 (contiguous K per column).
// Weight tile staged in LDS once per block. Each wave computes a 32-row x 128-col strip
// (two 16-row tiles sharing every B fragment). K templated so the k-loop fully unrolls:
// all A fragments preloaded in one clustered burst; B fragments double-buffered so the
// ds_load for fragment i+1 issues before the WMMAs consuming fragment i.
template <int K>
__global__ __launch_bounds__(256)
void hgt_gemm_bf16(const unsigned short* __restrict__ A, int M,
                   const unsigned short* __restrict__ WT,
                   const float* __restrict__ bias,
                   float* __restrict__ outF,
                   unsigned short* __restrict__ outB,
                   const float* __restrict__ xOld,
                   const float* __restrict__ skipPtr,
                   int mode) {
    constexpr int KS = K / 32;               // k-steps
    __shared__ unsigned short sW[Cc * K];

    // cooperative stage of WT[128][K] into LDS (uint4 granularity)
    constexpr int nvec = (Cc * K) >> 3;
#pragma unroll 2
    for (int i = threadIdx.x; i < nvec; i += 256)
        ((uint4*)sW)[i] = ((const uint4*)WT)[i];
    __syncthreads();

    const int lane = threadIdx.x & 31;
    const int wv   = threadIdx.x >> 5;
    const int tile = blockIdx.x * 8 + wv;    // 32-row macro-tile per wave
    const int row0 = tile * 32;
    if (row0 >= M) return;                   // wave-uniform: EXEC all-ones for WMMA
    const bool t1ok = (row0 + 16) < M;
    const int mlane = lane & 15;
    const int hi    = lane >> 4;

    const unsigned short* arow0 = A + (size_t)(row0 + mlane) * K;
    const unsigned short* arow1 = t1ok ? (arow0 + (size_t)16 * K) : arow0;

    // ---- preload ALL A fragments (KS k-steps x 2 row-tiles) in one burst ----
    Frag a[KS][2];
#pragma unroll
    for (int ks = 0; ks < KS; ++ks) {
        const int k0 = ks * 32;
        a[ks][0].q[0] = *(const uint4*)(arow0 + k0 + hi * 16);
        a[ks][0].q[1] = *(const uint4*)(arow0 + k0 + hi * 16 + 8);
        a[ks][1].q[0] = *(const uint4*)(arow1 + k0 + hi * 16);
        a[ks][1].q[1] = *(const uint4*)(arow1 + k0 + hi * 16 + 8);
    }

    v8f acc[2][8];
#pragma unroll
    for (int t = 0; t < 2; ++t) {
#pragma unroll
        for (int i = 0; i < 8; ++i) {
#pragma unroll
            for (int r = 0; r < 8; ++r) acc[t][i][r] = 0.0f;
        }
    }

    // B fragment address for (ks, nt): col = nt*16+mlane, chunk [ks*32 + hi*16, +16)
    const unsigned short* sBase = sW + (size_t)mlane * K + hi * 16;

    // ---- software-pipelined B fragments over the fully-unrolled (ks, nt) space ----
    Frag bc, bn;
    bc.q[0] = *(const uint4*)(sBase);
    bc.q[1] = *(const uint4*)(sBase + 8);
#pragma unroll
    for (int ks = 0; ks < KS; ++ks) {
#pragma unroll
        for (int nt = 0; nt < 8; ++nt) {
            const bool haveNext = (nt < 7) || (ks < KS - 1);
            if (haveNext) {
                const int nks = (nt < 7) ? ks : ks + 1;
                const int nnt = (nt < 7) ? nt + 1 : 0;
                const unsigned short* wrow = sBase + nnt * 16 * K + nks * 32;
                bn.q[0] = *(const uint4*)(wrow);
                bn.q[1] = *(const uint4*)(wrow + 8);
            }
            acc[0][nt] = __builtin_amdgcn_wmma_f32_16x16x32_bf16(
                false, a[ks][0].v, false, bc.v, (short)0, acc[0][nt], false, false);
            acc[1][nt] = __builtin_amdgcn_wmma_f32_16x16x32_bf16(
                false, a[ks][1].v, false, bc.v, (short)0, acc[1][nt], false, false);
            if (haveNext) bc.v = bn.v;
        }
    }

    float sg = 1.0f, cg = 0.0f;
    if (mode == GM_SKIP) {
        float sv = skipPtr[0];
        sg = 1.0f / (1.0f + __expf(-sv));
        cg = 1.0f - sg;
    }
#pragma unroll
    for (int t = 0; t < 2; ++t) {
        if (t == 1 && !t1ok) break;
        const int rbase = row0 + t * 16;
#pragma unroll
        for (int nt = 0; nt < 8; ++nt) {
            const int col = nt * 16 + mlane;
            const float b = bias ? bias[col] : 0.0f;
#pragma unroll
            for (int r = 0; r < 8; ++r) {
                const int row   = rbase + r + hi * 8;   // C layout: VGPR r -> M=r (+8 hi lanes)
                const size_t ix = (size_t)row * Cc + col;
                float v = acc[t][nt][r] + b;
                if (mode == GM_RELU)      v = v > 0.0f ? v : 0.0f;
                else if (mode == GM_SKIP) v = sg * v + cg * xOld[ix];
                outF[ix] = v;
                if (outB) outB[ix] = f2bf(v);
            }
        }
    }
}

static inline void launch_gemm(int K, int blocks, hipStream_t stream,
                               const unsigned short* A, int M,
                               const unsigned short* WT, const float* bias,
                               float* outF, unsigned short* outB,
                               const float* xOld, const float* skipPtr, int mode) {
    if (K == 128)
        hgt_gemm_bf16<128><<<blocks, 256, 0, stream>>>(A, M, WT, bias, outF, outB, xOld, skipPtr, mode);
    else if (K == 64)
        hgt_gemm_bf16<64><<<blocks, 256, 0, stream>>>(A, M, WT, bias, outF, outB, xOld, skipPtr, mode);
    else
        hgt_gemm_bf16<32><<<blocks, 256, 0, stream>>>(A, M, WT, bias, outF, outB, xOld, skipPtr, mode);
}

// ---------------- edge phase ----------------
// One wave per edge; lane owns 4 consecutive channels; 8 lanes per head.
__global__ __launch_bounds__(256)
void hgt_edge_logits(const int* __restrict__ src, const int* __restrict__ dst,
                     const float* __restrict__ Kf, const float* __restrict__ Qf,
                     const float* __restrict__ prel,
                     float* __restrict__ logits, unsigned* __restrict__ maxb, int E) {
    const int e = blockIdx.x * 8 + (threadIdx.x >> 5);
    if (e >= E) return;
    const int lane = threadIdx.x & 31;
    const int s = src[e], d = dst[e];
    const float4 q4 = ((const float4*)(Qf + (size_t)d * Cc))[lane];
    const float4 k4 = ((const float4*)(Kf + (size_t)s * Cc))[lane];
    float p = q4.x * k4.x + q4.y * k4.y + q4.z * k4.z + q4.w * k4.w;
    p += __shfl_xor(p, 1, 32);
    p += __shfl_xor(p, 2, 32);
    p += __shfl_xor(p, 4, 32);
    const int h = lane >> 3;
    if ((lane & 7) == 0) {
        const float lg = p * prel[h] * INV_SQRT_Df;
        logits[(size_t)e * Hh + h] = lg;
        atomicMax(maxb + (size_t)d * Hh + h, fkey(lg));
    }
}

__global__ __launch_bounds__(256)
void hgt_edge_accum(const int* __restrict__ src, const int* __restrict__ dst,
                    const float* __restrict__ Vf, const float* __restrict__ logits,
                    const unsigned* __restrict__ maxb, float* __restrict__ sumb,
                    float* __restrict__ agg, int E) {
    const int e = blockIdx.x * 8 + (threadIdx.x >> 5);
    if (e >= E) return;
    const int lane = threadIdx.x & 31;
    const int s = src[e], d = dst[e];
    const int h = lane >> 3;
    const float lg = logits[(size_t)e * Hh + h];
    const float m  = fdec(maxb[(size_t)d * Hh + h]);
    const float w  = expf(lg - m);
    if ((lane & 7) == 0) atomicAdd(sumb + (size_t)d * Hh + h, w);
    const float4 v4 = ((const float4*)(Vf + (size_t)s * Cc))[lane];
    float* o = agg + (size_t)d * Cc + lane * 4;
    atomicAdd(o + 0, w * v4.x);
    atomicAdd(o + 1, w * v4.y);
    atomicAdd(o + 2, w * v4.z);
    atomicAdd(o + 3, w * v4.w);
}

// normalize by softmax sum + exact GELU -> bf16 for the a_lin GEMM
__global__ __launch_bounds__(256)
void hgt_finalize_gelu(const float* __restrict__ agg, const float* __restrict__ sumb,
                       unsigned short* __restrict__ g, int n) {
    int i = blockIdx.x * 256 + threadIdx.x;
    if (i >= n) return;
    int node = i >> 7, c = i & 127, h = c >> 5;
    float s = sumb[(size_t)node * Hh + h];
    float x = agg[i] / (s + 1e-16f);
    float r = 0.5f * x * (1.0f + erff(x * 0.70710678118654752f));
    g[i] = f2bf(r);
}

// ---------------- host orchestration ----------------
extern "C" void kernel_launch(void* const* d_in, const int* in_sizes, int n_in,
                              void* d_out, int out_size, void* d_ws, size_t ws_size,
                              hipStream_t stream) {
    const float* x_a     = (const float*)d_in[0];
    const float* x_b     = (const float*)d_in[1];
    const int*   e_ab    = (const int*)d_in[2];
    const int*   e_ba    = (const int*)d_in[3];
    const float* lin_a_w = (const float*)d_in[4];
    const float* lin_a_b = (const float*)d_in[5];
    const float* lin_b_w = (const float*)d_in[6];
    const float* lin_b_b = (const float*)d_in[7];
    const float* k_w     = (const float*)d_in[8];
    const float* k_b     = (const float*)d_in[9];
    const float* q_w     = (const float*)d_in[10];
    const float* q_b     = (const float*)d_in[11];
    const float* v_w     = (const float*)d_in[12];
    const float* v_b     = (const float*)d_in[13];
    const float* a_rel   = (const float*)d_in[14];
    const float* m_rel   = (const float*)d_in[15];
    const float* p_rel   = (const float*)d_in[16];
    const float* a_lin_w = (const float*)d_in[17];
    const float* a_lin_b = (const float*)d_in[18];
    const float* skip    = (const float*)d_in[19];

    const int NA = in_sizes[0] / 64;
    const int NB = in_sizes[1] / 32;
    const int E  = in_sizes[2] / 2;

    // ---- carve workspace ----
    char*  base = (char*)d_ws;
    size_t off  = 0;
    auto carve = [&](size_t bytes) -> char* {
        char* r = base + off;
        off = (off + bytes + 255) & ~(size_t)255;
        return r;
    };
    float* xa = (float*)carve((size_t)NA * Cc * 4);
    float* xb = (float*)carve((size_t)NB * Cc * 4);
    unsigned short* xa_bf   = (unsigned short*)carve((size_t)NA * Cc * 2);
    unsigned short* xb_bf   = (unsigned short*)carve((size_t)NB * Cc * 2);
    unsigned short* xain_bf = (unsigned short*)carve((size_t)NA * 64 * 2);
    unsigned short* xbin_bf = (unsigned short*)carve((size_t)NB * 32 * 2);
    float* qa = (float*)carve((size_t)NA * Cc * 4);
    float* ka = (float*)carve((size_t)NA * Cc * 4);
    float* va = (float*)carve((size_t)NA * Cc * 4);
    float* qb = (float*)carve((size_t)NB * Cc * 4);
    float* kb = (float*)carve((size_t)NB * Cc * 4);
    float* vb = (float*)carve((size_t)NB * Cc * 4);
    float* agg_a = (float*)carve((size_t)NA * Cc * 4);
    float* agg_b = (float*)carve((size_t)NB * Cc * 4);
    unsigned short* g_a = (unsigned short*)carve((size_t)NA * Cc * 2);
    unsigned short* g_b = (unsigned short*)carve((size_t)NB * Cc * 2);
    float* logit_ab = (float*)carve((size_t)E * Hh * 4);
    float* logit_ba = (float*)carve((size_t)E * Hh * 4);
    unsigned* max_a = (unsigned*)carve((size_t)NA * Hh * 4);
    unsigned* max_b = (unsigned*)carve((size_t)NB * Hh * 4);
    float* sum_a = (float*)carve((size_t)NA * Hh * 4);
    float* sum_b = (float*)carve((size_t)NB * Hh * 4);
    unsigned short* linaT = (unsigned short*)carve((size_t)Cc * 64 * 2);
    unsigned short* linbT = (unsigned short*)carve((size_t)Cc * 32 * 2);
    unsigned short *WqT[4], *WkT[4], *WvT[4], *WaT[4];
    float *kbc[4], *vbc[4];
    for (int lt = 0; lt < 4; ++lt) {
        WqT[lt] = (unsigned short*)carve((size_t)Cc * Cc * 2);
        WkT[lt] = (unsigned short*)carve((size_t)Cc * Cc * 2);
        WvT[lt] = (unsigned short*)carve((size_t)Cc * Cc * 2);
        WaT[lt] = (unsigned short*)carve((size_t)Cc * Cc * 2);
        kbc[lt] = (float*)carve(Cc * 4);
        vbc[lt] = (float*)carve(Cc * 4);
    }

    // ---- weight preparation ----
    hgt_transpose_bf16<<<(64 * Cc + 255) / 256, 256, 0, stream>>>(lin_a_w, linaT, 64, Cc);
    hgt_transpose_bf16<<<(32 * Cc + 255) / 256, 256, 0, stream>>>(lin_b_w, linbT, 32, Cc);
    const int cgrid = (Cc * Cc + 255) / 256;
    for (int lt = 0; lt < 4; ++lt) {
        hgt_transpose_bf16<<<cgrid, 256, 0, stream>>>(q_w + (size_t)lt * Cc * Cc, WqT[lt], Cc, Cc);
        hgt_transpose_bf16<<<cgrid, 256, 0, stream>>>(a_lin_w + (size_t)lt * Cc * Cc, WaT[lt], Cc, Cc);
        hgt_combine_rel<<<cgrid, 256, 0, stream>>>(k_w + (size_t)lt * Cc * Cc, k_b + (size_t)lt * Cc,
                                                   a_rel + (size_t)lt * Hh * Dd * Dd, WkT[lt], kbc[lt]);
        hgt_combine_rel<<<cgrid, 256, 0, stream>>>(v_w + (size_t)lt * Cc * Cc, v_b + (size_t)lt * Cc,
                                                   m_rel + (size_t)lt * Hh * Dd * Dd, WvT[lt], vbc[lt]);
    }

    // ---- inputs -> bf16, input linear + relu ----
    hgt_cvt_bf16<<<(NA * 64 + 255) / 256, 256, 0, stream>>>(x_a, xain_bf, NA * 64);
    hgt_cvt_bf16<<<(NB * 32 + 255) / 256, 256, 0, stream>>>(x_b, xbin_bf, NB * 32);

    const int gBlkA = ((NA + 31) / 32 + 7) / 8;   // 32-row macro tiles, 8 waves/block
    const int gBlkB = ((NB + 31) / 32 + 7) / 8;
    launch_gemm(64, gBlkA, stream, xain_bf, NA, linaT, lin_a_b, xa, xa_bf, nullptr, nullptr, GM_RELU);
    launch_gemm(32, gBlkB, stream, xbin_bf, NB, linbT, lin_b_b, xb, xb_bf, nullptr, nullptr, GM_RELU);

    const int eBlk = (E + 7) / 8;
    for (int l = 0; l < Ll; ++l) {
        const int ltA = l * 2 + 0, ltB = l * 2 + 1;
        // Q, K', V' projections (rel transforms pre-folded into K'/V')
        launch_gemm(Cc, gBlkA, stream, xa_bf, NA, WqT[ltA], q_b + (size_t)ltA * Cc,
                    qa, nullptr, nullptr, nullptr, GM_BIAS);
        launch_gemm(Cc, gBlkA, stream, xa_bf, NA, WkT[ltA], kbc[ltA],
                    ka, nullptr, nullptr, nullptr, GM_BIAS);
        launch_gemm(Cc, gBlkA, stream, xa_bf, NA, WvT[ltA], vbc[ltA],
                    va, nullptr, nullptr, nullptr, GM_BIAS);
        launch_gemm(Cc, gBlkB, stream, xb_bf, NB, WqT[ltB], q_b + (size_t)ltB * Cc,
                    qb, nullptr, nullptr, nullptr, GM_BIAS);
        launch_gemm(Cc, gBlkB, stream, xb_bf, NB, WkT[ltB], kbc[ltB],
                    kb, nullptr, nullptr, nullptr, GM_BIAS);
        launch_gemm(Cc, gBlkB, stream, xb_bf, NB, WvT[ltB], vbc[ltB],
                    vb, nullptr, nullptr, nullptr, GM_BIAS);

        // edge type 0: A -> B (dst side B). memset(0) is identity for fkey-max & sums.
        hipMemsetAsync(max_b, 0, (size_t)NB * Hh * 4, stream);
        hipMemsetAsync(sum_b, 0, (size_t)NB * Hh * 4, stream);
        hipMemsetAsync(agg_b, 0, (size_t)NB * Cc * 4, stream);
        hgt_edge_logits<<<eBlk, 256, 0, stream>>>(e_ab, e_ab + E, ka, qb,
                                                  p_rel + (size_t)ltA * Hh, logit_ab, max_b, E);
        hgt_edge_accum<<<eBlk, 256, 0, stream>>>(e_ab, e_ab + E, va, logit_ab,
                                                 max_b, sum_b, agg_b, E);
        hgt_finalize_gelu<<<(NB * Cc + 255) / 256, 256, 0, stream>>>(agg_b, sum_b, g_b, NB * Cc);

        // edge type 1: B -> A (dst side A)
        hipMemsetAsync(max_a, 0, (size_t)NA * Hh * 4, stream);
        hipMemsetAsync(sum_a, 0, (size_t)NA * Hh * 4, stream);
        hipMemsetAsync(agg_a, 0, (size_t)NA * Cc * 4, stream);
        hgt_edge_logits<<<eBlk, 256, 0, stream>>>(e_ba, e_ba + E, kb, qa,
                                                  p_rel + (size_t)ltB * Hh, logit_ba, max_a, E);
        hgt_edge_accum<<<eBlk, 256, 0, stream>>>(e_ba, e_ba + E, vb, logit_ba,
                                                 max_a, sum_a, agg_a, E);
        hgt_finalize_gelu<<<(NA * Cc + 255) / 256, 256, 0, stream>>>(agg_a, sum_a, g_a, NA * Cc);

        // a_lin + sigmoid-gated skip. Last layer writes straight into d_out.
        const bool last = (l == Ll - 1);
        float* outA = last ? (float*)d_out : xa;
        float* outB = last ? ((float*)d_out + (size_t)NA * Cc) : xb;
        launch_gemm(Cc, gBlkA, stream, g_a, NA, WaT[ltA], a_lin_b + (size_t)ltA * Cc,
                    outA, last ? nullptr : xa_bf, xa, skip + ltA, GM_SKIP);
        launch_gemm(Cc, gBlkB, stream, g_b, NB, WaT[ltB], a_lin_b + (size_t)ltB * Cc,
                    outB, last ? nullptr : xb_bf, xb, skip + ltB, GM_SKIP);
    }
    (void)ws_size; (void)out_size; (void)n_in;
}
